// RNNDec_86930138071196
// MI455X (gfx1250) — compile-verified, matrix operand order checked by reading, not run on hardware
//
#include <hip/hip_runtime.h>
#include <cstddef>

// ---------------------------------------------------------------------------
// GRU decoder scan for MI455X (gfx1250, wave32, WMMA, TDM).
//   B=32, T=512, D=1024, 3D=3072
// Phase 1: Gi = X @ W_ih^T for ALL timesteps (bf16 WMMA, 2x2 register tiles).
// Phase 2: 512 stream-ordered step kernels: W_hh slice + h staged into LDS
//          via TENSOR_LOAD_TO_LDS (TDM), gh = h @ W_hh^T from LDS fragments,
//          gate combine in LDS, double-buffered h.
// ---------------------------------------------------------------------------

typedef __attribute__((ext_vector_type(16))) __bf16 v16bf;
typedef __attribute__((ext_vector_type(8)))  __bf16 v8bf;
typedef __attribute__((ext_vector_type(8)))  float  v8f;
typedef __attribute__((ext_vector_type(4)))  unsigned int u32x4;
typedef __attribute__((ext_vector_type(8)))  int i32x8;
typedef __attribute__((ext_vector_type(4)))  int i32x4;

#define BB 32
#define TT 512
#define DD 1024
#define D3 3072

// 16x32 bf16 WMMA operand fragment (ISA 7.12.2):
//   lanes 0-15 : row = lane,    K = {k0..k0+7}, {k0+16..k0+23}
//   lanes 16-31: row = lane-16, K = {k0+8..k0+15}, {k0+24..k0+31}
__device__ __forceinline__ v16bf load_frag16x32(const __bf16* base, int stride,
                                                int row0, int k0, int lane) {
    const int r  = lane & 15;
    const int ks = k0 + ((lane >> 4) << 3);
    const __bf16* p = base + (size_t)(row0 + r) * stride + ks;
    v8bf lo = *reinterpret_cast<const v8bf*>(p);
    v8bf hi = *reinterpret_cast<const v8bf*>(p + 16);
    v16bf out;
#pragma unroll
    for (int i = 0; i < 8; ++i) { out[i] = lo[i]; out[i + 8] = hi[i]; }
    return out;
}

// ------------------------- TDM descriptor (D#) ------------------------------
// 2D tile load: rows x cols elements of 2-byte data, row stride in elements.
// D# group0: count=1, lds_addr, global_addr, type=2. group1: data_size=1 (2B),
// tensor_dim0/1, tile_dim0/1, tensor_dim0_stride. groups 2-3 unused (zero).
// Toolchain: 6-arg builtin (u32x4, i32x8, i32x4, i32x4, i32x8, i32 cpol).
__device__ __forceinline__ void tdm_load_2d_bf16(unsigned lds_byte_addr,
                                                 const void* gptr,
                                                 unsigned rows, unsigned cols,
                                                 unsigned row_stride) {
    unsigned long long ga = (unsigned long long)(uintptr_t)gptr;
    u32x4 g0;
    g0[0] = 1u;                                           // count=1 (valid)
    g0[1] = lds_byte_addr;                                // lds_addr
    g0[2] = (unsigned)(ga & 0xFFFFFFFFu);                 // global_addr[31:0]
    g0[3] = (unsigned)((ga >> 32) & 0x01FFFFFFu) | (2u << 30); // [56:32] | type=2
    i32x8 g1;
    g1[0] = (int)(1u << 16);                              // data_size=1 (2 bytes)
    g1[1] = (int)((cols & 0xFFFFu) << 16);                // tensor_dim0[15:0]
    g1[2] = (int)(((cols >> 16) & 0xFFFFu) | ((rows & 0xFFFFu) << 16)); // dim0 hi | dim1 lo
    g1[3] = (int)(((rows >> 16) & 0xFFFFu) | ((cols & 0xFFFFu) << 16)); // dim1 hi | tile_dim0
    g1[4] = (int)(rows & 0xFFFFu);                        // tile_dim1 (tile_dim2=0)
    g1[5] = (int)row_stride;                              // tensor_dim0_stride[31:0]
    g1[6] = 0;                                            // stride hi | dim1_stride lo
    g1[7] = 0;
    i32x4 gz4 = {0, 0, 0, 0};
    i32x8 gz8 = {0, 0, 0, 0, 0, 0, 0, 0};
    __builtin_amdgcn_tensor_load_to_lds(g0, g1, gz4, gz4, gz8, 0);
}

// ------------------------- utility kernels ---------------------------------

__global__ void f32_to_bf16_kernel(const float* __restrict__ src,
                                   __bf16* __restrict__ dst, int n) {
    int i = blockIdx.x * blockDim.x + threadIdx.x;
    if (i < n) dst[i] = (__bf16)src[i];
}

__global__ void zero_h_kernel(float* __restrict__ hf, __bf16* __restrict__ hb, int n) {
    int i = blockIdx.x * blockDim.x + threadIdx.x;
    if (i < n) { hf[i] = 0.0f; hb[i] = (__bf16)0.0f; }
}

// ------------------------- Phase 1: Gi GEMM --------------------------------
// Gi[m, n] = sum_k X[m,k] * W_ih[n,k]. Each wave computes a 32x32 output
// block (2x2 tiles of 16x16) -> 4 independent WMMAs per K-step, 2x operand
// reuse in registers.

__global__ void gemm_gi_kernel(const __bf16* __restrict__ Xbf,
                               const __bf16* __restrict__ Wihbf,
                               float* __restrict__ Gi) {
    const int lane = threadIdx.x & 31;
    const int wave = blockIdx.x * (blockDim.x >> 5) + (threadIdx.x >> 5);
    const int NT2 = D3 / 32;             // 96 N-blocks
    const int m0 = (wave / NT2) * 32;
    const int n0 = (wave % NT2) * 32;

    v8f acc00 = {}, acc01 = {}, acc10 = {}, acc11 = {};
#pragma unroll 2
    for (int k = 0; k < DD; k += 32) {
        v16bf a0 = load_frag16x32(Xbf,   DD, m0,      k, lane);
        v16bf a1 = load_frag16x32(Xbf,   DD, m0 + 16, k, lane);
        v16bf b0 = load_frag16x32(Wihbf, DD, n0,      k, lane);
        v16bf b1 = load_frag16x32(Wihbf, DD, n0 + 16, k, lane);
        acc00 = __builtin_amdgcn_wmma_f32_16x16x32_bf16(false, a0, false, b0, (short)0, acc00, false, false);
        acc01 = __builtin_amdgcn_wmma_f32_16x16x32_bf16(false, a0, false, b1, (short)0, acc01, false, false);
        acc10 = __builtin_amdgcn_wmma_f32_16x16x32_bf16(false, a1, false, b0, (short)0, acc10, false, false);
        acc11 = __builtin_amdgcn_wmma_f32_16x16x32_bf16(false, a1, false, b1, (short)0, acc11, false, false);
    }

    // C layout: lane -> column (lane&15); VGPR j -> row j (+8 for lanes>=16)
    const int c  = lane & 15;
    const int rb = (lane >> 4) << 3;
    v8f* accs[4] = {&acc00, &acc01, &acc10, &acc11};
#pragma unroll
    for (int q = 0; q < 4; ++q) {
        const int mt = q >> 1, nt = q & 1;
        float* dst = Gi + (size_t)(m0 + mt * 16 + rb) * D3 + (n0 + nt * 16 + c);
#pragma unroll
        for (int j = 0; j < 8; ++j) dst[(size_t)j * D3] = (*accs[q])[j];
    }
}

// ------------------------- Phase 2: GRU step -------------------------------
// 64 workgroups x 192 threads (6 waves). Workgroup owns 16 h-columns.
// TDM stages the 3-gate W_hh slice (3x16x1024) and h (32x1024) into LDS.
// Wave w: gate = w/2 (r,z,n), mtile = w&1. gh tiles staged in LDS, then
// gate combine + h update + output store.

struct alignas(16) StepLds {
    __bf16 W[3][16][DD];   // 96 KB: W_hh rows for r/z/n column slice
    __bf16 H[BB][DD];      // 64 KB: h_{t-1} (bf16)
    float  gh[3][BB][16];  //  6 KB: partial gh tiles
};

__global__ void gru_step_kernel(const float* __restrict__ Gi,
                                const __bf16* __restrict__ Whhbf,
                                const float* __restrict__ b_ih,
                                const float* __restrict__ b_hh,
                                const float* __restrict__ h_in_f,
                                const __bf16* __restrict__ h_in_bf,
                                float* __restrict__ h_out_f,
                                __bf16* __restrict__ h_out_bf,
                                float* __restrict__ out, int t) {
    __shared__ StepLds lds;

    const int lane  = threadIdx.x & 31;
    const int wave  = threadIdx.x >> 5;   // 0..5
    const int c0    = blockIdx.x * 16;    // column slice of D

    // Wave 0 issues the TDM loads; TENSORcnt is per-wave, so it waits and
    // everyone synchronizes at the barrier.
    if (wave == 0) {
#pragma unroll
        for (int g = 0; g < 3; ++g) {
            tdm_load_2d_bf16((unsigned)(offsetof(StepLds, W) + (size_t)g * 16 * DD * 2),
                             Whhbf + (size_t)(g * DD + c0) * DD,
                             16u, (unsigned)DD, (unsigned)DD);
        }
        tdm_load_2d_bf16((unsigned)offsetof(StepLds, H), h_in_bf,
                         (unsigned)BB, (unsigned)DD, (unsigned)DD);
        __builtin_amdgcn_s_wait_tensorcnt(0);
    }
    __syncthreads();

    const int gate  = wave >> 1;          // 0=r, 1=z, 2=n
    const int mtile = wave & 1;

    v8f acc = {};
    {
        const int r  = lane & 15;
        const int ko = (lane >> 4) << 3;
        const __bf16* arow = &lds.H[mtile * 16 + r][0];
        const __bf16* brow = &lds.W[gate][r][0];
#pragma unroll 4
        for (int k = 0; k < DD; k += 32) {
            v8bf alo = *reinterpret_cast<const v8bf*>(arow + k + ko);
            v8bf ahi = *reinterpret_cast<const v8bf*>(arow + k + ko + 16);
            v8bf blo = *reinterpret_cast<const v8bf*>(brow + k + ko);
            v8bf bhi = *reinterpret_cast<const v8bf*>(brow + k + ko + 16);
            v16bf a, b;
#pragma unroll
            for (int i = 0; i < 8; ++i) {
                a[i] = alo[i]; a[i + 8] = ahi[i];
                b[i] = blo[i]; b[i + 8] = bhi[i];
            }
            acc = __builtin_amdgcn_wmma_f32_16x16x32_bf16(
                      false, a, false, b, (short)0, acc, false, false);
        }
    }

    const int col   = lane & 15;
    const int rbase = mtile * 16 + ((lane >> 4) << 3);
#pragma unroll
    for (int j = 0; j < 8; ++j)
        lds.gh[gate][rbase + j][col] = acc[j];

    __syncthreads();

    // Gate combine: 32x16 = 512 elements over 192 threads.
    for (int e = threadIdx.x; e < BB * 16; e += blockDim.x) {
        const int b = e >> 4;
        const int c = e & 15;
        const int d = c0 + c;
        const size_t gi_row = ((size_t)b * TT + t) * D3;

        float ir  = Gi[gi_row + d]          + b_ih[d];
        float iz  = Gi[gi_row + DD + d]     + b_ih[DD + d];
        float in_ = Gi[gi_row + 2 * DD + d] + b_ih[2 * DD + d];
        float hr  = lds.gh[0][b][c] + b_hh[d];
        float hz  = lds.gh[1][b][c] + b_hh[DD + d];
        float hn  = lds.gh[2][b][c] + b_hh[2 * DD + d];

        float r = 1.0f / (1.0f + __expf(-(ir + hr)));
        float z = 1.0f / (1.0f + __expf(-(iz + hz)));
        float n = tanhf(in_ + r * hn);

        float hprev = h_in_f[b * DD + d];
        float hnew  = (1.0f - z) * n + z * hprev;

        h_out_f[b * DD + d]  = hnew;
        h_out_bf[b * DD + d] = (__bf16)hnew;
        out[((size_t)b * TT + t) * DD + d] = hnew;
    }
}

// ------------------------------ launcher -----------------------------------

extern "C" void kernel_launch(void* const* d_in, const int* in_sizes, int n_in,
                              void* d_out, int out_size, void* d_ws, size_t ws_size,
                              hipStream_t stream) {
    const float* h_enc = (const float*)d_in[0];   // [B,T,D]
    const float* W_ih  = (const float*)d_in[1];   // [3D,D]
    const float* W_hh  = (const float*)d_in[2];   // [3D,D]
    const float* b_ih  = (const float*)d_in[3];   // [3D]
    const float* b_hh  = (const float*)d_in[4];   // [3D]
    float* out = (float*)d_out;                   // [B,T,D]

    // Workspace layout (all offsets 256B aligned)
    char* ws = (char*)d_ws;
    const size_t nX = (size_t)BB * TT * DD;       // 16,777,216
    const size_t nW = (size_t)D3 * DD;            //  3,145,728
    __bf16* Xbf   = (__bf16*)(ws);                // 33,554,432 B
    __bf16* Wihbf = (__bf16*)(ws + 33554432);     //  6,291,456 B
    __bf16* Whhbf = (__bf16*)(ws + 39845888);     //  6,291,456 B
    float*  Gi    = (float*) (ws + 46137344);     // 201,326,592 B
    float*  hF0   = (float*) (ws + 247463936);    // 131,072 B
    float*  hF1   = (float*) (ws + 247595008);    // 131,072 B
    __bf16* hB0   = (__bf16*)(ws + 247726080);    //  65,536 B
    __bf16* hB1   = (__bf16*)(ws + 247791616);    //  65,536 B

    // fp32 -> bf16 conversions
    f32_to_bf16_kernel<<<(int)((nX + 255) / 256), 256, 0, stream>>>(h_enc, Xbf, (int)nX);
    f32_to_bf16_kernel<<<(int)((nW + 255) / 256), 256, 0, stream>>>(W_ih, Wihbf, (int)nW);
    f32_to_bf16_kernel<<<(int)((nW + 255) / 256), 256, 0, stream>>>(W_hh, Whhbf, (int)nW);

    // h0 = 0 (buffer 0 is read at t=0)
    zero_h_kernel<<<(BB * DD + 255) / 256, 256, 0, stream>>>(hF0, hB0, BB * DD);

    // Phase 1: Gi = X @ W_ih^T. 512*96 waves of 32x32 blocks, 8 waves/block.
    {
        const int total_waves = (BB * TT / 32) * (D3 / 32);  // 49,152
        gemm_gi_kernel<<<total_waves / 8, 256, 0, stream>>>(Xbf, Wihbf, Gi);
    }

    // Phase 2: 512 dependent steps, double-buffered h.
    float*  hf[2] = {hF0, hF1};
    __bf16* hb[2] = {hB0, hB1};
    for (int t = 0; t < TT; ++t) {
        const int src = t & 1, dst = src ^ 1;
        gru_step_kernel<<<DD / 16, 192, 0, stream>>>(
            Gi, Whhbf, b_ih, b_hh,
            hf[src], hb[src], hf[dst], hb[dst], out, t);
    }
}
